// GAT_72988674228319
// MI455X (gfx1250) — matrix operand order repaired and checked by reading, not compiled, not run on hardware
//
#include <hip/hip_runtime.h>
#include <hip/hip_bf16.h>

typedef __attribute__((ext_vector_type(16))) _Float16 v16h;
typedef __attribute__((ext_vector_type(8)))  float    v8f;

#define NNODES 100000
#define NEDGES 800000
#define NETOT  900000   /* edges + self loops */
#define ROWT   6250     /* NNODES / 16 */

// ---------------------------------------------------------------------------
// Pack float matrix rows into CDNA5 WMMA 16-bit A-operand layout (16x32 tiles)
// Apack[rowTile][kChunk(8)][lane(32)*16 halves]
// A layout (ISA 7.12.2): lanes 0-15 rows M=0..15, VGPR v -> K = 2v(+8 if v>=4),
// lanes 16-31 same rows, K += 8.
// ---------------------------------------------------------------------------
__global__ void pack_a_f16(const float* __restrict__ src, _Float16* __restrict__ dst,
                           int rowTiles, int ldsrc) {
  int idx = blockIdx.x * blockDim.x + threadIdx.x;
  int total = rowTiles * 8 * 512;
  if (idx >= total) return;
  int r    = idx >> 12;        // row tile
  int rem  = idx & 4095;
  int c    = rem >> 9;         // k-chunk
  int q    = rem & 511;
  int lane = q >> 4;
  int j    = q & 15;
  int v = j >> 1, half = j & 1;
  int kloc = 2 * v + (v >= 4 ? 8 : 0) + ((lane & 16) ? 8 : 0) + half;
  int M = r * 16 + (lane & 15);
  int K = c * 32 + kloc;
  dst[idx] = (_Float16)src[(size_t)M * ldsrc + K];
}

// ---------------------------------------------------------------------------
// Pack weight matrix [K, ncols] into WMMA 16-bit B-operand layout (32x16 tiles)
// B layout: VGPR v lanes 0-15 hold K=2v,2v+1 (+8 if v>=4); lanes 16-31 K += 8.
// Columns >= ncols are zero-padded.
// ---------------------------------------------------------------------------
__global__ void pack_b_f16(const float* __restrict__ W, _Float16* __restrict__ dst,
                           int colTiles, int ncols) {
  int idx = blockIdx.x * blockDim.x + threadIdx.x;
  int total = colTiles * 8 * 512;
  if (idx >= total) return;
  int t    = idx >> 12;
  int rem  = idx & 4095;
  int c    = rem >> 9;
  int q    = rem & 511;
  int lane = q >> 4;
  int j    = q & 15;
  int v = j >> 1, half = j & 1;
  int kloc = 2 * v + (v >= 4 ? 8 : 0) + half + ((lane & 16) ? 8 : 0);
  int n = t * 16 + (lane & 15);
  int K = c * 32 + kloc;
  float val = (n < ncols) ? W[(size_t)K * ncols + n] : 0.0f;
  dst[idx] = (_Float16)val;
}

// ---------------------------------------------------------------------------
// Layer-1 GEMM: block(256)=8 waves owns one 16-row tile x all 16 col tiles.
// A tile (8KB) staged once per block into LDS with the gfx1250 async
// global->LDS path; each wave keeps 2 column-tile accumulators in registers
// (no spills). B streams from the L2-resident 128KB packed weights.
// ---------------------------------------------------------------------------
__global__ void __launch_bounds__(256) gemm1_wmma(const _Float16* __restrict__ Apack,
                                                  const _Float16* __restrict__ Bpack,
                                                  float* __restrict__ C, int ldc) {
  __shared__ _Float16 smA[4096];
  int rowTile = blockIdx.x;
  int wave = threadIdx.x >> 5;
  int lane = threadIdx.x & 31;

  // async stage: each thread moves 16B of the A row-tile into LDS.
  // The LDS operand is derived from the real smA pointer (ptrtoint => the
  // array escapes, so the asm's memory clobber legally writes it), and the
  // unprovable guarded store below keeps LDS loads from folding to poison.
  {
    unsigned long long gaddr =
        (unsigned long long)(Apack + (size_t)rowTile * 4096 + threadIdx.x * 8);
    unsigned ldsOff =
        (unsigned)(unsigned long long)(smA + (size_t)threadIdx.x * 8);
    if (gaddr == 0) smA[0] = (_Float16)0.0f;   // never true; not provably so
    asm volatile("global_load_async_to_lds_b128 %0, %1, off"
                 :: "v"(ldsOff), "v"(gaddr) : "memory");
  }
  asm volatile("s_wait_asynccnt 0x0" ::: "memory");
  __syncthreads();

  v8f acc0 = {}, acc1 = {};
  const int t0 = wave * 2, t1 = wave * 2 + 1;
  const _Float16* bP = Bpack + lane * 16;
#pragma unroll
  for (int c = 0; c < 8; ++c) {
    v16h a  = *(const v16h*)(smA + c * 512 + lane * 16);        // ds_load_b128 x2
    v16h b0 = *(const v16h*)(bP + ((size_t)t0 * 8 + c) * 512);
    v16h b1 = *(const v16h*)(bP + ((size_t)t1 * 8 + c) * 512);
    acc0 = __builtin_amdgcn_wmma_f32_16x16x32_f16(false, a, false, b0, (short)0,
                                                  acc0, false, false);
    acc1 = __builtin_amdgcn_wmma_f32_16x16x32_f16(false, a, false, b1, (short)0,
                                                  acc1, false, false);
  }
  // C/D layout: lane -> col = t*16 + lane%16; VGPR v -> row v (+8 if lane>=16)
  int colLo = lane & 15;
  int rOff  = (lane & 16) ? 8 : 0;
#pragma unroll
  for (int v = 0; v < 8; ++v) {
    size_t row = (size_t)(rowTile * 16 + v + rOff);
    C[row * ldc + t0 * 16 + colLo] = acc0[v];
    C[row * ldc + t1 * 16 + colLo] = acc1[v];
  }
}

// ---------------------------------------------------------------------------
// Small-N GEMM (layer 2): one wave per 16-row tile, NT col tiles in registers.
// ---------------------------------------------------------------------------
template <int NT>
__global__ void gemm_wmma_f16(const _Float16* __restrict__ Apack,
                              const _Float16* __restrict__ Bpack,
                              float* __restrict__ C, int rowTiles, int ldc) {
  int gid  = blockIdx.x * blockDim.x + threadIdx.x;
  int wave = gid >> 5;
  int lane = gid & 31;
  if (wave >= rowTiles) return;          // wave-uniform: EXEC stays all-1s

  v8f acc[NT] = {};
  const _Float16* aP = Apack + (size_t)wave * 4096 + lane * 16;
  const _Float16* bP = Bpack + lane * 16;
#pragma unroll
  for (int c = 0; c < 8; ++c) {
    v16h a = *(const v16h*)(aP + c * 512);
#pragma unroll
    for (int t = 0; t < NT; ++t) {
      v16h b = *(const v16h*)(bP + ((size_t)t * 8 + c) * 512);
      acc[t] = __builtin_amdgcn_wmma_f32_16x16x32_f16(false, a, false, b, (short)0,
                                                      acc[t], false, false);
    }
  }
  int colLo = lane & 15;
  int rOff  = (lane & 16) ? 8 : 0;
#pragma unroll
  for (int t = 0; t < NT; ++t) {
#pragma unroll
    for (int v = 0; v < 8; ++v) {
      C[(size_t)(wave * 16 + v + rOff) * ldc + t * 16 + colLo] = acc[t][v];
    }
  }
}

// a_src[n,h] = sum_c feat[n,h,c]*attS[h,c] ; same for a_dst
__global__ void attn_coef(const float* __restrict__ feat, const float* __restrict__ attS,
                          const float* __restrict__ attD, float* __restrict__ aS,
                          float* __restrict__ aD, int n, int Hh, int Cc, int ldRow) {
  int i = blockIdx.x * blockDim.x + threadIdx.x;
  if (i >= n * Hh) return;
  int node = i / Hh, hd = i % Hh;
  const float* row = feat + (size_t)node * ldRow + hd * Cc;
  float s = 0.0f, d = 0.0f;
  for (int c = 0; c < Cc; ++c) {
    float v = row[c];
    s += v * attS[hd * Cc + c];
    d += v * attD[hd * Cc + c];
  }
  aS[i] = s;
  aD[i] = d;
}

// monotone float<->uint map for atomicMax on floats
__device__ __forceinline__ unsigned flipf(float f) {
  unsigned u = __float_as_uint(f);
  return (u & 0x80000000u) ? ~u : (u | 0x80000000u);
}
__device__ __forceinline__ float unflipf(unsigned u) {
  return __uint_as_float((u & 0x80000000u) ? (u & 0x7FFFFFFFu) : ~u);
}

__device__ __forceinline__ void edge_sd(int e, int nE, const int* __restrict__ srcI,
                                        const int* __restrict__ dstI, int& s, int& d) {
  if (e < nE) { s = srcI[e]; d = dstI[e]; } else { s = e - nE; d = s; }  // self loop
}

// pass 1: segment max of leaky-relu logits (flipped-uint atomicMax)
template <int HH>
__global__ void edge_max(const int* __restrict__ srcI, const int* __restrict__ dstI,
                         const float* __restrict__ aS, const float* __restrict__ aD,
                         unsigned* __restrict__ mFlip, int nE, int nTot) {
  int i = blockIdx.x * blockDim.x + threadIdx.x;
  if (i >= nTot * HH) return;
  int e = i / HH, hd = i % HH;
  int s, d; edge_sd(e, nE, srcI, dstI, s, d);
  float l = aS[s * HH + hd] + aD[d * HH + hd];
  l = (l > 0.0f) ? l : l * 0.2f;
  atomicMax(&mFlip[d * HH + hd], flipf(l));
}

// pass 2: ex = exp(logit - m[dst]); denom[dst] += ex
template <int HH>
__global__ void edge_expsum(const int* __restrict__ srcI, const int* __restrict__ dstI,
                            const float* __restrict__ aS, const float* __restrict__ aD,
                            const unsigned* __restrict__ mFlip, float* __restrict__ ex,
                            float* __restrict__ denom, int nE, int nTot) {
  int i = blockIdx.x * blockDim.x + threadIdx.x;
  if (i >= nTot * HH) return;
  int e = i / HH, hd = i % HH;
  int s, d; edge_sd(e, nE, srcI, dstI, s, d);
  float l = aS[s * HH + hd] + aD[d * HH + hd];
  l = (l > 0.0f) ? l : l * 0.2f;
  float v = __expf(l - unflipf(mFlip[d * HH + hd]));
  ex[i] = v;
  atomicAdd(&denom[d * HH + hd], v);
}

// pass 3: out[dst] += alpha * feat[src]   (one thread = 4 channels; feat L2-hot)
template <int HH, int CC>
__global__ void edge_agg(const int* __restrict__ srcI, const int* __restrict__ dstI,
                         const float* __restrict__ ex, const float* __restrict__ denom,
                         const float* __restrict__ feat, float* __restrict__ out,
                         int nE, int nTot, int ldf, int ldo) {
  constexpr int GROUPS = (HH * CC) >> 2;
  int i = blockIdx.x * blockDim.x + threadIdx.x;
  if (i >= nTot * GROUPS) return;
  int e = i / GROUPS, j = i % GROUPS;
  int ch = j * 4;
  int hd = ch / CC;
  int s, d; edge_sd(e, nE, srcI, dstI, s, d);
  float alpha = ex[(size_t)e * HH + hd] / (denom[d * HH + hd] + 1e-16f);
  float4 hv = *(const float4*)(feat + (size_t)s * ldf + ch);
  float* op = out + (size_t)d * ldo + ch;
  atomicAdd(op + 0, alpha * hv.x);
  atomicAdd(op + 1, alpha * hv.y);
  atomicAdd(op + 2, alpha * hv.z);
  atomicAdd(op + 3, alpha * hv.w);
}

__global__ void bias_elu(const float* __restrict__ acc, const float* __restrict__ bias,
                         float* __restrict__ out, int n, int F) {
  int i = blockIdx.x * blockDim.x + threadIdx.x;
  if (i >= n * F) return;
  float v = acc[i] + bias[i % F];
  out[i] = (v > 0.0f) ? v : (__expf(v) - 1.0f);
}

__global__ void out_bias_init(const float* __restrict__ bias, float* __restrict__ out,
                              int n, int F) {
  int i = blockIdx.x * blockDim.x + threadIdx.x;
  if (i >= n * F) return;
  out[i] = bias[i % F];
}

static inline int cdiv(long long a, int b) { return (int)((a + b - 1) / b); }

extern "C" void kernel_launch(void* const* d_in, const int* in_sizes, int n_in,
                              void* d_out, int out_size, void* d_ws, size_t ws_size,
                              hipStream_t stream) {
  (void)in_sizes; (void)n_in; (void)out_size; (void)ws_size;
  const float* x     = (const float*)d_in[0];
  const int*   ei    = (const int*)d_in[1];
  const float* W1    = (const float*)d_in[2];
  const float* attS1 = (const float*)d_in[3];
  const float* attD1 = (const float*)d_in[4];
  const float* b1    = (const float*)d_in[5];
  const float* W2    = (const float*)d_in[6];
  const float* attS2 = (const float*)d_in[7];
  const float* attD2 = (const float*)d_in[8];
  const float* b2    = (const float*)d_in[9];
  float* out = (float*)d_out;

  const int* srcI = ei;
  const int* dstI = ei + NEDGES;

  // ---- workspace layout -------------------------------------------------
  char* ws = (char*)d_ws;
  size_t off = 0;
  auto carve = [&](size_t bytes) -> char* {
    char* p = ws + off;
    off = (off + bytes + 255) & ~(size_t)255;
    return p;
  };
  _Float16* Apack = (_Float16*)carve((size_t)NNODES * 256 * 2);  // reused for h2 pack
  _Float16* B1p   = (_Float16*)carve((size_t)16 * 4096 * 2);
  _Float16* B2p   = (_Float16*)carve((size_t)3 * 4096 * 2);
  float*    h1    = (float*)carve((size_t)NNODES * 256 * 4);
  float*    a1s   = (float*)carve((size_t)NNODES * 8 * 4);
  float*    a1d   = (float*)carve((size_t)NNODES * 8 * 4);
  unsigned* m1    = (unsigned*)carve((size_t)NNODES * 8 * 4);
  float*    den1  = (float*)carve((size_t)NNODES * 8 * 4);
  float*    ex1   = (float*)carve((size_t)NETOT * 8 * 4);
  float*    acc1  = (float*)carve((size_t)NNODES * 256 * 4);   // becomes h2 after ELU
  float*    g2    = (float*)carve((size_t)NNODES * 48 * 4);    // padded ldc=48
  float*    a2s   = (float*)carve((size_t)NNODES * 4);
  float*    a2d   = (float*)carve((size_t)NNODES * 4);
  unsigned* m2    = (unsigned*)carve((size_t)NNODES * 4);
  float*    den2  = (float*)carve((size_t)NNODES * 4);
  float*    ex2   = (float*)carve((size_t)NETOT * 4);

  const int B = 256;

  // ---- layer 1 ----------------------------------------------------------
  pack_a_f16<<<cdiv((long long)ROWT * 4096, B), B, 0, stream>>>(x, Apack, ROWT, 256);
  pack_b_f16<<<cdiv(16 * 4096, B), B, 0, stream>>>(W1, B1p, 16, 256);
  pack_b_f16<<<cdiv(3 * 4096, B), B, 0, stream>>>(W2, B2p, 3, 40);

  gemm1_wmma<<<ROWT, B, 0, stream>>>(Apack, B1p, h1, 256);

  attn_coef<<<cdiv((long long)NNODES * 8, B), B, 0, stream>>>(h1, attS1, attD1, a1s, a1d,
                                                              NNODES, 8, 32, 256);

  hipMemsetAsync(m1, 0, (size_t)NNODES * 8 * 4, stream);     // flip-space -inf
  edge_max<8><<<cdiv((long long)NETOT * 8, B), B, 0, stream>>>(srcI, dstI, a1s, a1d, m1,
                                                               NEDGES, NETOT);
  hipMemsetAsync(den1, 0, (size_t)NNODES * 8 * 4, stream);
  edge_expsum<8><<<cdiv((long long)NETOT * 8, B), B, 0, stream>>>(srcI, dstI, a1s, a1d,
                                                                  m1, ex1, den1,
                                                                  NEDGES, NETOT);
  hipMemsetAsync(acc1, 0, (size_t)NNODES * 256 * 4, stream);
  edge_agg<8, 32><<<cdiv((long long)NETOT * 64, B), B, 0, stream>>>(srcI, dstI, ex1, den1,
                                                                    h1, acc1, NEDGES,
                                                                    NETOT, 256, 256);
  bias_elu<<<cdiv((long long)NNODES * 256, B), B, 0, stream>>>(acc1, b1, acc1, NNODES, 256);

  // ---- layer 2 ----------------------------------------------------------
  pack_a_f16<<<cdiv((long long)ROWT * 4096, B), B, 0, stream>>>(acc1, Apack, ROWT, 256);
  gemm_wmma_f16<3><<<cdiv((long long)ROWT * 32, B), B, 0, stream>>>(Apack, B2p, g2, ROWT, 48);

  attn_coef<<<cdiv(NNODES, B), B, 0, stream>>>(g2, attS2, attD2, a2s, a2d, NNODES, 1, 40, 48);

  hipMemsetAsync(m2, 0, (size_t)NNODES * 4, stream);
  edge_max<1><<<cdiv(NETOT, B), B, 0, stream>>>(srcI, dstI, a2s, a2d, m2, NEDGES, NETOT);
  hipMemsetAsync(den2, 0, (size_t)NNODES * 4, stream);
  edge_expsum<1><<<cdiv(NETOT, B), B, 0, stream>>>(srcI, dstI, a2s, a2d, m2, ex2, den2,
                                                   NEDGES, NETOT);
  out_bias_init<<<cdiv((long long)NNODES * 40, B), B, 0, stream>>>(b2, out, NNODES, 40);
  edge_agg<1, 40><<<cdiv((long long)NETOT * 10, B), B, 0, stream>>>(srcI, dstI, ex2, den2,
                                                                    g2, out, NEDGES,
                                                                    NETOT, 48, 40);
}
